// TemporalGraph_17033840296512
// MI455X (gfx1250) — matrix operand order repaired and checked by reading, not compiled, not run on hardware
//
#include <hip/hip_runtime.h>
#include <hip/hip_bf16.h>

typedef __attribute__((ext_vector_type(16))) _Float16 v16h;
typedef __attribute__((ext_vector_type(8)))  _Float16 v8h;
typedef __attribute__((ext_vector_type(8)))  float    v8f;

#define NODES 22
#define BATCH 256
#define TLEN  500
#define KP    512                  // padded K / padded out-col count
#define M_TOT (BATCH * NODES)      // 5632 = 352 * 16 (exact 16-row tiles)
#define EMB   40
#define HOPS  3

// ---------------- ws layout (bytes) ----------------
// [0)                      A   : 484 floats (softmax(adj))
// [2048)                   A2  : 484 floats (A @ A)
// [4096)                   H   : 3 hops * 5632 * 512 f16   (17,301,504 B)
// [4096 + Hbytes)          WT  : 3 hops * 512 * 512 f16 (W transposed, padded)
#define WS_H_OFF  4096ull
#define WS_WT_OFF (WS_H_OFF + (unsigned long long)HOPS * M_TOT * KP * 2ull)

// ---------- kernel 1: softmax rows of adj, and A2 = A @ A ----------
__global__ void prep_adj_kernel(const float* __restrict__ adj,
                                float* __restrict__ Aout,
                                float* __restrict__ A2out) {
    __shared__ float As[NODES * NODES];
    int tid = threadIdx.x;  // 64 threads
    if (tid < NODES) {
        float row[NODES];
        float mx = -3.4e38f;
        for (int m = 0; m < NODES; ++m) {
            row[m] = adj[tid * NODES + m];
            mx = fmaxf(mx, row[m]);
        }
        float s = 0.f;
        for (int m = 0; m < NODES; ++m) { row[m] = __expf(row[m] - mx); s += row[m]; }
        float inv = 1.0f / s;
        for (int m = 0; m < NODES; ++m) {
            float v = row[m] * inv;
            As[tid * NODES + m] = v;
            Aout[tid * NODES + m] = v;
        }
    }
    __syncthreads();
    for (int i = tid; i < NODES * NODES; i += 64) {
        int n = i / NODES, j = i - n * NODES;
        float s = 0.f;
        for (int m = 0; m < NODES; ++m) s += As[n * NODES + m] * As[m * NODES + j];
        A2out[i] = s;
    }
}

// ---------- kernel 2: H0 = x, H1 = A x, H2 = A^2 x  (f16, K padded to 512) ----------
__global__ __launch_bounds__(256) void build_H_kernel(const float* __restrict__ x,
                                                      const float* __restrict__ Aw,
                                                      const float* __restrict__ A2w,
                                                      _Float16* __restrict__ H) {
    __shared__ float As[NODES * NODES];
    __shared__ float A2s[NODES * NODES];
    int tid = threadIdx.x;
    for (int i = tid; i < NODES * NODES; i += 256) { As[i] = Aw[i]; A2s[i] = A2w[i]; }
    __syncthreads();

    int g = blockIdx.x * 256 + tid;        // over BATCH * 512
    int b = g >> 9;
    int t = g & (KP - 1);

    _Float16* H0 = H;
    _Float16* H1 = H + (size_t)M_TOT * KP;
    _Float16* H2 = H + 2ull * M_TOT * KP;
    size_t rowbase = (size_t)b * NODES * KP + t;

    if (t >= TLEN) {  // zero pad columns 500..511
        for (int n = 0; n < NODES; ++n) {
            H0[rowbase + (size_t)n * KP] = (_Float16)0.f;
            H1[rowbase + (size_t)n * KP] = (_Float16)0.f;
            H2[rowbase + (size_t)n * KP] = (_Float16)0.f;
        }
        return;
    }
    float xv[NODES];
    for (int n = 0; n < NODES; ++n)
        xv[n] = x[((size_t)b * NODES + n) * TLEN + t];
    for (int n = 0; n < NODES; ++n) {
        float s1 = 0.f, s2 = 0.f;
        for (int m = 0; m < NODES; ++m) {
            s1 += As[n * NODES + m] * xv[m];
            s2 += A2s[n * NODES + m] * xv[m];
        }
        H0[rowbase + (size_t)n * KP] = (_Float16)xv[n];
        H1[rowbase + (size_t)n * KP] = (_Float16)s1;
        H2[rowbase + (size_t)n * KP] = (_Float16)s2;
    }
}

// ---------- kernel 3: WT[h][n][k] = f16(W[h][k][n]), zero-padded to 512x512 ----------
__global__ __launch_bounds__(256) void build_WT_kernel(const float* __restrict__ W,
                                                       _Float16* __restrict__ WT) {
    int g = blockIdx.x * 256 + threadIdx.x;   // over HOPS * 512 * 512
    int h = g >> 18;
    int n = (g >> 9) & (KP - 1);
    int k = g & (KP - 1);
    float v = 0.f;
    if (k < TLEN && n < TLEN)
        v = W[(size_t)h * TLEN * TLEN + (size_t)k * TLEN + n];
    WT[((size_t)h * KP + n) * KP + k] = (_Float16)v;
}

// ---------- kernel 4: fused WMMA GEMM (3 hops) + bias + 40x broadcast ----------
__global__ __launch_bounds__(256) void gemm_bcast_kernel(const _Float16* __restrict__ H,
                                                         const _Float16* __restrict__ WT,
                                                         const float* __restrict__ bias,
                                                         float* __restrict__ out) {
    const int lane = threadIdx.x & 31;
    const int wave = threadIdx.x >> 5;   // 8 waves
    const int wy   = wave >> 1;          // 0..3  -> M sub-tile
    const int wx   = wave & 1;           // 0..1  -> N half
    const int m0   = blockIdx.y * 64 + wy * 16;
    const int n0   = blockIdx.x * 128 + wx * 64;
    const int r    = lane & 15;
    const int hl   = lane >> 4;

    v8f acc[4] = {v8f{}, v8f{}, v8f{}, v8f{}};

    for (int h = 0; h < HOPS; ++h) {
        const _Float16* Hh = H  + (size_t)h * M_TOT * KP;
        const _Float16* Wh = WT + (size_t)h * KP * KP;
        // A fragment rows: lanes 0..15 and 16..31 both index M = m0 + r;
        // half selects K offset (+16), second v8h covers K+8.
        const _Float16* arow = Hh + (size_t)(m0 + r) * KP + hl * 16;
        // B fragment: lane = N column, half selects K+16; contiguous 16 f16.
        const _Float16* brow0 = Wh + (size_t)(n0 +  0 + r) * KP + hl * 16;
        const _Float16* brow1 = Wh + (size_t)(n0 + 16 + r) * KP + hl * 16;
        const _Float16* brow2 = Wh + (size_t)(n0 + 32 + r) * KP + hl * 16;
        const _Float16* brow3 = Wh + (size_t)(n0 + 48 + r) * KP + hl * 16;

        #pragma unroll 4
        for (int k0 = 0; k0 < KP; k0 += 32) {
            union { v16h v; v8h half2[2]; } af;
            af.half2[0] = *(const v8h*)(arow + k0);
            af.half2[1] = *(const v8h*)(arow + k0 + 8);
            v16h bf0 = *(const v16h*)(brow0 + k0);
            v16h bf1 = *(const v16h*)(brow1 + k0);
            v16h bf2 = *(const v16h*)(brow2 + k0);
            v16h bf3 = *(const v16h*)(brow3 + k0);
            acc[0] = __builtin_amdgcn_wmma_f32_16x16x32_f16(false, af.v, false, bf0,
                                                            (short)0, acc[0], false, false);
            acc[1] = __builtin_amdgcn_wmma_f32_16x16x32_f16(false, af.v, false, bf1,
                                                            (short)0, acc[1], false, false);
            acc[2] = __builtin_amdgcn_wmma_f32_16x16x32_f16(false, af.v, false, bf2,
                                                            (short)0, acc[2], false, false);
            acc[3] = __builtin_amdgcn_wmma_f32_16x16x32_f16(false, af.v, false, bf3,
                                                            (short)0, acc[3], false, false);
        }
    }

    // Epilogue: C/D layout -> lane holds (M = m0 + 8*hl + p, N = n0t + r) for p=0..7.
    #pragma unroll
    for (int t = 0; t < 4; ++t) {
        int c = n0 + t * 16 + r;
        if (c < TLEN) {
            float bv = bias[c];
            #pragma unroll
            for (int p = 0; p < 8; ++p) {
                int m  = m0 + hl * 8 + p;
                int bb = m / NODES;
                int nn = m - bb * NODES;
                float val = acc[t][p] + bv;
                float* dst = out + ((size_t)bb * EMB * NODES + nn) * TLEN + c;
                #pragma unroll 8
                for (int e = 0; e < EMB; ++e) {
                    *dst = val;
                    dst += (size_t)NODES * TLEN;
                }
            }
        }
    }
}

extern "C" void kernel_launch(void* const* d_in, const int* in_sizes, int n_in,
                              void* d_out, int out_size, void* d_ws, size_t ws_size,
                              hipStream_t stream) {
    const float* x    = (const float*)d_in[0];   // (256, 22, 500)
    const float* adj  = (const float*)d_in[1];   // (22, 22)
    const float* W    = (const float*)d_in[2];   // (3, 500, 500)
    const float* bias = (const float*)d_in[3];   // (500,)
    float* out        = (float*)d_out;           // (256, 40, 22, 500)

    float*    Aw  = (float*)d_ws;
    float*    A2w = (float*)((char*)d_ws + 2048);
    _Float16* H   = (_Float16*)((char*)d_ws + WS_H_OFF);
    _Float16* WT  = (_Float16*)((char*)d_ws + WS_WT_OFF);

    prep_adj_kernel<<<1, 64, 0, stream>>>(adj, Aw, A2w);
    build_WT_kernel<<<(HOPS * KP * KP) / 256, 256, 0, stream>>>(W, WT);
    build_H_kernel<<<(BATCH * KP) / 256, 256, 0, stream>>>(x, Aw, A2w, H);
    gemm_bcast_kernel<<<dim3(KP / 128, M_TOT / 64), 256, 0, stream>>>(H, WT, bias, out);
}
